// Lvl1VQ_79843442032955
// MI455X (gfx1250) — matrix operand order repaired and checked by reading, not compiled, not run on hardware
//
#include <hip/hip_runtime.h>

// Problem constants (match reference)
#define B_ 16
#define D_ 64
#define L_ 8192
#define K_ 512

#define MT 64            // positions (l) per workgroup
#define KC 128           // codebook codes per LDS chunk
#define NCHUNK (K_ / KC) // 4
#define DP 72            // padded ushort stride per code row in LDS:
                         // 144B => every fragment base 16B-aligned (ds_load_b128),
                         // start banks = distinct multiples of 4 across 16 lanes

typedef __attribute__((ext_vector_type(16))) __bf16 v16bf;
typedef __attribute__((ext_vector_type(8)))  float  v8f;

union FragBF {
    v16bf v;
    unsigned short u[16];
    uint4 q[2];          // 2 x 16B for clean ds_load_b128 from LDS
};

__device__ __forceinline__ unsigned short bf16_rne(float f) {
    unsigned u = __builtin_bit_cast(unsigned, f);
    u += 0x7FFFu + ((u >> 16) & 1u);            // round-to-nearest-even
    return (unsigned short)(u >> 16);
}
__device__ __forceinline__ float bf16_to_f32(unsigned short h) {
    unsigned u = ((unsigned)h) << 16;
    return __builtin_bit_cast(float, u);
}

__global__ __launch_bounds__(256)
void vq_argmin_gather_kernel(const float* __restrict__ zg,   // [B][D][L]
                             const float* __restrict__ cbg,  // [K][D]
                             float* __restrict__ out_idx,    // [B][L] (as f32)
                             float* __restrict__ out_zq,     // [B][D][L]
                             float* __restrict__ out_al,     // [B][D][L]
                             float* __restrict__ out_cm)     // [B][D][L]
{
    __shared__ __align__(16) unsigned short s_cb_hi[KC * DP];
    __shared__ __align__(16) unsigned short s_cb_lo[KC * DP];
    __shared__ float s_c2[KC];
    __shared__ float s_part[256];
    __shared__ float s_bval[2][MT];
    __shared__ int   s_bidx[2][MT];
    __shared__ int   s_idx[MT];

    const int tid  = threadIdx.x;
    const int lane = tid & 31;
    const int wave = tid >> 5;
    const int wm   = wave & 3;   // M-subtile (16 positions each)
    const int wn   = wave >> 2;  // N-group: codes [wn*64, wn*64+64) within chunk
    const int wg   = blockIdx.x;
    const int b    = wg / (L_ / MT);
    const int l0   = (wg % (L_ / MT)) * MT;

    const float* zb = zg + (size_t)b * D_ * L_;

    // ---- Build A fragments (z tile as bf16 hi/lo), reused for all code chunks ----
    // A 16x32 layout (contiguous, matches B): lanes 0-15 hold K=0..15; lanes 16-31 hold K=16..31
    FragBF a_hi[2], a_lo[2];
    {
        const int mrow   = l0 + wm * 16 + (lane & 15);
        const int hiHalf = lane >> 4;
        for (int kh = 0; kh < 2; ++kh) {
            #pragma unroll
            for (int i = 0; i < 16; ++i) {
                int dl;
                dl = hiHalf * 16 + i;
                const int d = kh * 32 + dl;
                const float f = zb[(size_t)d * L_ + mrow];
                const unsigned short h = bf16_rne(f);
                a_hi[kh].u[i] = h;
                a_lo[kh].u[i] = bf16_rne(f - bf16_to_f32(h));
            }
        }
    }

    float best[8];
    int   bidx[8];
    #pragma unroll
    for (int v = 0; v < 8; ++v) { best[v] = 3.4e38f; bidx[v] = 0; }

    for (int chunk = 0; chunk < NCHUNK; ++chunk) {
        __syncthreads();  // previous chunk's compute done before LDS overwrite
        // ---- Stage KC codes: convert f32 -> bf16 hi/lo into LDS; exact-f32 |c|^2 ----
        {
            const int kl = tid >> 1;            // 0..127 local code
            const int dh = (tid & 1) * 32;      // half of D
            const float* crow = cbg + (size_t)(chunk * KC + kl) * D_ + dh;
            float ssq = 0.f;
            #pragma unroll 8
            for (int j = 0; j < 32; ++j) {
                const float f = crow[j];
                ssq += f * f;
                const unsigned short h = bf16_rne(f);
                s_cb_hi[kl * DP + dh + j] = h;
                s_cb_lo[kl * DP + dh + j] = bf16_rne(f - bf16_to_f32(h));
            }
            s_part[tid] = ssq;
            if (chunk + 1 < NCHUNK) {
                // hide next chunk's codebook latency (global_prefetch_b8)
                __builtin_prefetch(cbg + (size_t)((chunk + 1) * KC + kl) * D_ + dh, 0, 1);
            }
        }
        __syncthreads();
        if (tid < KC) s_c2[tid] = s_part[2 * tid] + s_part[2 * tid + 1];
        __syncthreads();

        // ---- GEMM + running argmin: wave covers 4 k-tiles of 16 codes ----
        #pragma unroll 2
        for (int kt = 0; kt < 4; ++kt) {
            const int ncol   = lane & 15;
            const int klocal = wn * 64 + kt * 16 + ncol;     // code within chunk (per-lane column)
            const int doff   = (lane >> 4) * 16;             // B K-split: lanes0-15 K=0..15, 16-31 K=16..31
            // Fragment bases are 16B aligned (DP=72): pure ds_load_b128, no repacking
            FragBF b_hi[2], b_lo[2];
            #pragma unroll
            for (int kh = 0; kh < 2; ++kh) {
                const int dbase = kh * 32 + doff;
                const uint4* ph = (const uint4*)&s_cb_hi[klocal * DP + dbase];
                const uint4* pl = (const uint4*)&s_cb_lo[klocal * DP + dbase];
                b_hi[kh].q[0] = ph[0]; b_hi[kh].q[1] = ph[1];
                b_lo[kh].q[0] = pl[0]; b_lo[kh].q[1] = pl[1];
            }
            v8f acc = {};
            #pragma unroll
            for (int kh = 0; kh < 2; ++kh) {
                // split-f32 product: zh*ch + zh*cl + zl*ch  (~f32 accuracy)
                acc = __builtin_amdgcn_wmma_f32_16x16x32_bf16(false, a_hi[kh].v, false, b_hi[kh].v, (short)0, acc, false, false);
                acc = __builtin_amdgcn_wmma_f32_16x16x32_bf16(false, a_hi[kh].v, false, b_lo[kh].v, (short)0, acc, false, false);
                acc = __builtin_amdgcn_wmma_f32_16x16x32_bf16(false, a_lo[kh].v, false, b_hi[kh].v, (short)0, acc, false, false);
            }
            // argmin over score = 0.5*||c||^2 - z.c  (||z||^2 constant per row).
            // Within a lane k is scanned ascending, so strict '<' keeps the first
            // minimum -- no index tie-break needed here (only in cross-lane merge).
            const float halfc2 = 0.5f * s_c2[klocal];
            const int   kglob  = chunk * KC + klocal;
            #pragma unroll
            for (int v = 0; v < 8; ++v) {
                const float score = halfc2 - acc[v];
                if (score < best[v]) { best[v] = score; bidx[v] = kglob; }
            }
        }
    }

    // ---- Cross-lane argmin within each 16-lane half (rows v and v+8) ----
    #pragma unroll
    for (int m = 8; m >= 1; m >>= 1) {
        #pragma unroll
        for (int v = 0; v < 8; ++v) {
            const float ov = __shfl_xor(best[v], m, 32);
            const int   oi = __shfl_xor(bidx[v], m, 32);
            if (ov < best[v] || (ov == best[v] && oi < bidx[v])) { best[v] = ov; bidx[v] = oi; }
        }
    }
    if ((lane & 15) == 0) {
        const int rbase = wm * 16 + ((lane >> 4) ? 8 : 0);
        #pragma unroll
        for (int v = 0; v < 8; ++v) {
            s_bval[wn][rbase + v] = best[v];
            s_bidx[wn][rbase + v] = bidx[v];
        }
    }
    __syncthreads();

    // ---- Combine the two N-groups, emit indices ----
    if (tid < MT) {
        const float v0 = s_bval[0][tid], v1 = s_bval[1][tid];
        const int   i0 = s_bidx[0][tid], i1 = s_bidx[1][tid];
        const int sel = (v1 < v0 || (v1 == v0 && i1 < i0)) ? i1 : i0;
        s_idx[tid] = sel;
        out_idx[(size_t)b * L_ + l0 + tid] = (float)sel;
    }
    __syncthreads();

    // ---- Gather exact-f32 codebook rows -> z_q, and losses (identical values) ----
    #pragma unroll 4
    for (int it = 0; it < (MT * D_) / 256; ++it) {   // 16 iterations
        const int e  = it * 256 + tid;
        const int d  = e >> 6;      // / MT
        const int ml = e & (MT - 1);
        const int idx = s_idx[ml];
        const size_t go = (size_t)b * D_ * L_ + (size_t)d * L_ + l0 + ml;
        const float c  = cbg[(size_t)idx * D_ + d];
        const float z  = zb[(size_t)d * L_ + l0 + ml];
        const float df = z - c;
        const float sq = df * df;
        out_zq[go] = c;
        out_al[go] = sq;
        out_cm[go] = sq;
    }
}

extern "C" void kernel_launch(void* const* d_in, const int* in_sizes, int n_in,
                              void* d_out, int out_size, void* d_ws, size_t ws_size,
                              hipStream_t stream) {
    (void)in_sizes; (void)n_in; (void)out_size; (void)d_ws; (void)ws_size;
    const float* zg  = (const float*)d_in[0];   // z_e [B,D,L] f32
    const float* cbg = (const float*)d_in[1];   // codebook [K,D] f32

    float* out_idx = (float*)d_out;                       // B*L
    float* out_zq  = out_idx + (size_t)B_ * L_;           // B*D*L
    float* out_al  = out_zq  + (size_t)B_ * D_ * L_;      // B*D*L
    float* out_cm  = out_al  + (size_t)B_ * D_ * L_;      // B*D*L

    const int grid = B_ * (L_ / MT);   // 2048 workgroups
    vq_argmin_gather_kernel<<<grid, 256, 0, stream>>>(zg, cbg, out_idx, out_zq, out_al, out_cm);
}
